// ChebyNet_20375324852684
// MI455X (gfx1250) — matrix operand order repaired and checked by reading, not compiled, not run on hardware
//
#include <hip/hip_runtime.h>

// ---------------------------------------------------------------------------
// ChebyNet (K=3) forward for MI455X / gfx1250, wave32 + WMMA f16->f32.
// ---------------------------------------------------------------------------

typedef __attribute__((ext_vector_type(16))) _Float16 v16h;
typedef __attribute__((ext_vector_type(8)))  _Float16 v8h;
typedef __attribute__((ext_vector_type(8)))  float    v8f;

#define NNODES 50000
#define NEDGES 800000
#define FDIM   100
#define HDIM   64
#define CDIM   40
#define KPAD1  320   // 3*100 padded to multiple of 32
#define KPAD2  192   // 3*64 (already multiple of 32)
#define NPAD2  48    // C=40 padded to multiple of 16

static __device__ __forceinline__ void atomic_add_f32(float* p, float v) {
    __hip_atomic_fetch_add(p, v, __ATOMIC_RELAXED, __HIP_MEMORY_SCOPE_AGENT);
}

// ---------------- graph normalization ----------------

__global__ void k_zero(float* __restrict__ p, int n) {
    int i = blockIdx.x * blockDim.x + threadIdx.x;
    if (i < n) p[i] = 0.0f;
}

__global__ void k_deg(const long long* __restrict__ row, float* __restrict__ deg, int E) {
    int e = blockIdx.x * blockDim.x + threadIdx.x;
    if (e < E) atomic_add_f32(&deg[(int)row[e]], 1.0f);
}

__global__ void k_dinv(float* __restrict__ d, int n) {
    int i = blockIdx.x * blockDim.x + threadIdx.x;
    if (i < n) {
        float v = d[i];
        d[i] = (v > 0.0f) ? rsqrtf(v) : 0.0f;
    }
}

__global__ void k_norm(const long long* __restrict__ row, const long long* __restrict__ col,
                       const float* __restrict__ dinv, float* __restrict__ nrm, int E) {
    int e = blockIdx.x * blockDim.x + threadIdx.x;
    if (e < E) nrm[e] = -(dinv[(int)row[e]] * dinv[(int)col[e]]);
}

__global__ void k_negcopy(float* __restrict__ dst, const float* __restrict__ src, int n) {
    int i = blockIdx.x * blockDim.x + threadIdx.x;
    if (i < n) dst[i] = -src[i];
}

// ---------------- SPMV: dst[row] += scale * norm[e] * src[col] ----------------
// Thread = (edge, 4-feature chunk). Fd is a multiple of 4 so float4 loads are
// 16B-aligned. Atomics land in L2 (entire dst fits in the 192MB L2).

__global__ void k_spmv(const float* __restrict__ src, float* __restrict__ dst,
                       const long long* __restrict__ row, const long long* __restrict__ col,
                       const float* __restrict__ nrm, float scale,
                       int Fd, int chunks, int total) {
    int t = blockIdx.x * blockDim.x + threadIdx.x;
    if (t >= total) return;
    int e = t / chunks;
    int c = (t - e * chunks) * 4;
    int r  = (int)row[e];
    int cc = (int)col[e];
    float w = scale * nrm[e];
    const float4 v = *(const float4*)(src + (size_t)cc * Fd + c);
    float* d = dst + (size_t)r * Fd + c;
    atomic_add_f32(d + 0, w * v.x);
    atomic_add_f32(d + 1, w * v.y);
    atomic_add_f32(d + 2, w * v.z);
    atomic_add_f32(d + 3, w * v.w);
}

// ---------------- f16 packing ----------------
// A: row-major [M x Kpad], K = [Tx0 | Tx1 | Tx2 | zero-pad], f16.
__global__ void k_packA(_Float16* __restrict__ A, const float* __restrict__ s0,
                        const float* __restrict__ s1, const float* __restrict__ s2,
                        int Fsrc, int Kpad, int total) {
    int t = blockIdx.x * blockDim.x + threadIdx.x;
    if (t >= total) return;
    int n = t / Kpad;
    int k = t - n * Kpad;
    int s = k / Fsrc;
    int kk = k - s * Fsrc;
    float v = 0.0f;
    size_t idx = (size_t)n * Fsrc + kk;
    if      (s == 0) v = s0[idx];
    else if (s == 1) v = s1[idx];
    else if (s == 2) v = s2[idx];
    A[t] = (_Float16)v;
}

// B: column-major f16 [Npad x Kpad] (Bc[n*Kpad + k] = W[s][kk][n]), zero-padded.
__global__ void k_packW(_Float16* __restrict__ Wc, const float* __restrict__ W,
                        int Fsrc, int Nout, int Kpad, int Npad) {
    int t = blockIdx.x * blockDim.x + threadIdx.x;
    int total = Npad * Kpad;
    if (t >= total) return;
    int n = t / Kpad;
    int k = t - n * Kpad;
    int s = k / Fsrc;
    int kk = k - s * Fsrc;
    float v = 0.0f;
    if (n < Nout && s < 3) v = W[((size_t)s * Fsrc + kk) * Nout + n];
    Wc[t] = (_Float16)v;
}

// ---------------- WMMA GEMM: out = A @ B + bias (optional ReLU) ----------------
// One wave per 16x16 output tile; wave w of a block covers output columns
// [16w, 16w+16). M is an exact multiple of 16 so EXEC==all-ones at every WMMA.
//
// Fragment layouts per CDNA5 ISA 7.12.2 (wave32):
//   A 16x32 f16:  lane L: row = L&15, g = L>>4;
//                 halves[0..7]  = K[k0 + 8g .. +7],
//                 halves[8..15] = K[k0 + 16 + 8g .. +7]          (2x b128 loads)
//   B 32x16 f16:  lane L: col = L&15, g = L>>4;
//                 halves[0..7]  = K[k0 + 8g .. +7],
//                 halves[8..15] = K[k0 + 16 + 8g .. +7]          (2x b128 loads)
//   C/D 16x16 f32: VGPR v: row = v + 8g, col = L&15.

__global__ void k_gemm_wmma(const _Float16* __restrict__ A, const _Float16* __restrict__ Bc,
                            const float* __restrict__ bias, float* __restrict__ out,
                            int Kpad, int Nout, int ldOut, int relu) {
    const int lane = threadIdx.x & 31;
    const int wv   = threadIdx.x >> 5;
    const int g    = lane >> 4;
    const int ml   = lane & 15;

    const _Float16* arow = A  + (size_t)(blockIdx.x * 16 + ml) * Kpad;
    const _Float16* bcol = Bc + (size_t)(wv * 16 + ml) * Kpad;

    v8f acc = {};
    for (int k0 = 0; k0 < Kpad; k0 += 32) {
        // hint the next A strip toward the caches (global_prefetch_b8)
        __builtin_prefetch(arow + k0 + 32, 0, 3);

        v8h a0 = *(const v8h*)(arow + k0 + 8 * g);
        v8h a1 = *(const v8h*)(arow + k0 + 16 + 8 * g);
        v8h b0 = *(const v8h*)(bcol + k0 + 8 * g);
        v8h b1 = *(const v8h*)(bcol + k0 + 16 + 8 * g);
        v16h af = __builtin_shufflevector(a0, a1, 0,1,2,3,4,5,6,7,8,9,10,11,12,13,14,15);
        v16h bf = __builtin_shufflevector(b0, b1, 0,1,2,3,4,5,6,7,8,9,10,11,12,13,14,15);
        acc = __builtin_amdgcn_wmma_f32_16x16x32_f16(
                  /*neg_a=*/false, af, /*neg_b=*/false, bf,
                  /*c_mod=*/(short)0, acc, /*reuse_a=*/false, /*reuse_b=*/false);
    }

    const int colg = wv * 16 + ml;
    if (colg < Nout) {
        const float bvv = bias[colg];
        const int rbase = blockIdx.x * 16 + 8 * g;
        #pragma unroll
        for (int v = 0; v < 8; ++v) {
            float x = acc[v] + bvv;
            if (relu) x = fmaxf(x, 0.0f);
            out[(size_t)(rbase + v) * ldOut + colg] = x;
        }
    }
}

// ---------------- log-softmax over C=40, one wave32 per row, in place ----------

__global__ void k_logsoftmax40(float* __restrict__ out, int M) {
    int w = (int)((blockIdx.x * (long long)blockDim.x + threadIdx.x) >> 5);
    int lane = threadIdx.x & 31;
    if (w >= M) return;
    float* r = out + (size_t)w * CDIM;

    float a = r[lane];                                       // cols 0..31
    float b = (lane < 8) ? r[32 + lane] : -__builtin_huge_valf(); // cols 32..39

    float m = fmaxf(a, b);
    #pragma unroll
    for (int off = 16; off > 0; off >>= 1) m = fmaxf(m, __shfl_xor(m, off, 32));

    float s = __expf(a - m) + ((lane < 8) ? __expf(b - m) : 0.0f);
    #pragma unroll
    for (int off = 16; off > 0; off >>= 1) s += __shfl_xor(s, off, 32);

    float ls = __logf(s);
    r[lane] = a - m - ls;
    if (lane < 8) r[32 + lane] = b - m - ls;
}

// ---------------------------------------------------------------------------

extern "C" void kernel_launch(void* const* d_in, const int* in_sizes, int n_in,
                              void* d_out, int out_size, void* d_ws, size_t ws_size,
                              hipStream_t stream) {
    (void)in_sizes; (void)n_in; (void)out_size; (void)ws_size;

    const float*     x   = (const float*)d_in[0];
    const long long* ei  = (const long long*)d_in[1];
    const float*     W1  = (const float*)d_in[2];
    const float*     b1  = (const float*)d_in[3];
    const float*     W2  = (const float*)d_in[4];
    const float*     b2  = (const float*)d_in[5];
    const long long* row = ei;
    const long long* col = ei + NEDGES;
    float* out = (float*)d_out;

    // workspace carve-up (256B aligned)
    char* base = (char*)d_ws;
    size_t off = 0;
    auto alloc = [&](size_t bytes) -> void* {
        void* p = base + off;
        off = (off + bytes + 255) & ~(size_t)255;
        return p;
    };
    float*     Dg  = (float*)    alloc((size_t)NNODES * 4);
    float*     NRM = (float*)    alloc((size_t)NEDGES * 4);
    float*     T1a = (float*)    alloc((size_t)NNODES * FDIM * 4);
    float*     T2a = (float*)    alloc((size_t)NNODES * FDIM * 4);
    _Float16*  A1  = (_Float16*) alloc((size_t)NNODES * KPAD1 * 2);
    _Float16*  Wc1 = (_Float16*) alloc((size_t)KPAD1 * HDIM * 2);
    float*     H1  = (float*)    alloc((size_t)NNODES * HDIM * 4);
    float*     T1b = (float*)    alloc((size_t)NNODES * HDIM * 4);
    float*     T2b = (float*)    alloc((size_t)NNODES * HDIM * 4);
    _Float16*  A2  = (_Float16*) alloc((size_t)NNODES * KPAD2 * 2);
    _Float16*  Wc2 = (_Float16*) alloc((size_t)KPAD2 * NPAD2 * 2);

    const int B = 256;
    auto gridFor = [](int n, int b) { return (n + b - 1) / b; };

    // ---- normalization: deg -> dinv -> per-edge norm
    k_zero<<<gridFor(NNODES, B), B, 0, stream>>>(Dg, NNODES);
    k_deg <<<gridFor(NEDGES, B), B, 0, stream>>>(row, Dg, NEDGES);
    k_dinv<<<gridFor(NNODES, B), B, 0, stream>>>(Dg, NNODES);
    k_norm<<<gridFor(NEDGES, B), B, 0, stream>>>(row, col, Dg, NRM, NEDGES);

    // ---- layer 1 Chebyshev bases (f32)
    const int nf = NNODES * FDIM;
    const int tot1 = NEDGES * (FDIM / 4);
    k_zero   <<<gridFor(nf, B), B, 0, stream>>>(T1a, nf);
    k_spmv   <<<gridFor(tot1, B), B, 0, stream>>>(x,   T1a, row, col, NRM, 1.0f, FDIM, FDIM / 4, tot1);
    k_negcopy<<<gridFor(nf, B), B, 0, stream>>>(T2a, x, nf);                 // T2a = -Tx0
    k_spmv   <<<gridFor(tot1, B), B, 0, stream>>>(T1a, T2a, row, col, NRM, 2.0f, FDIM, FDIM / 4, tot1);

    // ---- layer 1 GEMM (K-concat WMMA) + bias + ReLU
    const int totA1 = NNODES * KPAD1;
    k_packA<<<gridFor(totA1, B), B, 0, stream>>>(A1, x, T1a, T2a, FDIM, KPAD1, totA1);
    k_packW<<<gridFor(KPAD1 * HDIM, B), B, 0, stream>>>(Wc1, W1, FDIM, HDIM, KPAD1, HDIM);
    k_gemm_wmma<<<NNODES / 16, 32 * (HDIM / 16), 0, stream>>>(A1, Wc1, b1, H1, KPAD1, HDIM, HDIM, 1);

    // ---- layer 2 Chebyshev bases (f32)
    const int nh = NNODES * HDIM;
    const int tot2 = NEDGES * (HDIM / 4);
    k_zero   <<<gridFor(nh, B), B, 0, stream>>>(T1b, nh);
    k_spmv   <<<gridFor(tot2, B), B, 0, stream>>>(H1,  T1b, row, col, NRM, 1.0f, HDIM, HDIM / 4, tot2);
    k_negcopy<<<gridFor(nh, B), B, 0, stream>>>(T2b, H1, nh);
    k_spmv   <<<gridFor(tot2, B), B, 0, stream>>>(T1b, T2b, row, col, NRM, 2.0f, HDIM, HDIM / 4, tot2);

    // ---- layer 2 GEMM -> logits straight into d_out
    const int totA2 = NNODES * KPAD2;
    k_packA<<<gridFor(totA2, B), B, 0, stream>>>(A2, H1, T1b, T2b, HDIM, KPAD2, totA2);
    k_packW<<<gridFor(KPAD2 * NPAD2, B), B, 0, stream>>>(Wc2, W2, HDIM, CDIM, KPAD2, NPAD2);
    k_gemm_wmma<<<NNODES / 16, 32 * (NPAD2 / 16), 0, stream>>>(A2, Wc2, b2, out, KPAD2, CDIM, CDIM, 0);

    // ---- log-softmax (wave per row, in place)
    k_logsoftmax40<<<NNODES / 8, 256, 0, stream>>>(out, NNODES);
}
